// KLECE_19292993094182
// MI455X (gfx1250) — matrix operand order repaired and checked by reading, not compiled, not run on hardware
//
#include <hip/hip_runtime.h>
#include <hip/hip_fp16.h>
#include <math.h>
#include <float.h>

#define N_COLS 100
#define NB 15

typedef __attribute__((ext_vector_type(16))) _Float16 v16h;
typedef __attribute__((ext_vector_type(8)))  float    v8f;

struct Ws {
  double totalSq;      // sum of sm^2 over all elements (f64 for 25M-term accuracy)
  float  sumF0[16];    // sum of sm[:,0] per floor-bin
  float  sumF1[16];    // sum of sm[:,1] per floor-bin
  int    cntC0[16];    // ceil-bin counts (table build), col 0
  int    cntC1[16];    // ceil-bin counts, col 1
  int    cntF0[16];    // floor-bin counts (lookup), col 0
  int    cntF1[16];    // floor-bin counts, col 1
};

__global__ void ece_init(Ws* __restrict__ ws) {
  int t = threadIdx.x;
  if (t == 0) ws->totalSq = 0.0;
  if (t < 16) {
    ws->sumF0[t] = 0.f; ws->sumF1[t] = 0.f;
    ws->cntC0[t] = 0;   ws->cntC1[t] = 0;
    ws->cntF0[t] = 0;   ws->cntF1[t] = 0;
  }
}

// One wave (32 lanes) per row. Lanes 0-24 hold the row as float4 (b128 loads,
// fully coalesced: row stride 400 B is 16B-aligned). Butterfly reductions for
// max and exp-sum; lane 0 owns columns 0..3 and feeds the LDS bin accumulators.
__global__ void __launch_bounds__(256) ece_pass(const float* __restrict__ in,
                                                Ws* __restrict__ ws, int nRows) {
  __shared__ int   sC0[16], sC1[16], sF0[16], sF1[16];
  __shared__ float sS0[16], sS1[16];
  __shared__ float sWave[8];
  const int tid = threadIdx.x;
  if (tid < 16) {
    sC0[tid] = 0; sC1[tid] = 0; sF0[tid] = 0; sF1[tid] = 0;
    sS0[tid] = 0.f; sS1[tid] = 0.f;
  }
  __syncthreads();

  const int lane      = tid & 31;
  const int waveInBlk = tid >> 5;
  const int wave      = blockIdx.x * 8 + waveInBlk;
  const int nWaves    = gridDim.x * 8;

  float accSq = 0.f;
  for (int row = wave; row < nRows; row += nWaves) {
    const float4* p = (const float4*)(in + (size_t)row * N_COLS);
    float x0, x1, x2, x3;
    if (lane < 25) {
      float4 v = p[lane];
      x0 = v.x; x1 = v.y; x2 = v.z; x3 = v.w;
    } else {
      x0 = x1 = x2 = x3 = -FLT_MAX;
    }
    float m = fmaxf(fmaxf(x0, x1), fmaxf(x2, x3));
    #pragma unroll
    for (int off = 16; off > 0; off >>= 1) m = fmaxf(m, __shfl_xor(m, off, 32));

    float e0 = 0.f, e1 = 0.f, e2 = 0.f, e3 = 0.f;
    if (lane < 25) {
      e0 = __expf(x0 - m); e1 = __expf(x1 - m);
      e2 = __expf(x2 - m); e3 = __expf(x3 - m);
    }
    float s = (e0 + e1) + (e2 + e3);
    #pragma unroll
    for (int off = 16; off > 0; off >>= 1) s += __shfl_xor(s, off, 32);

    float inv = 1.0f / s;
    float p0 = e0 * inv, p1 = e1 * inv, p2 = e2 * inv, p3 = e3 * inv;
    accSq += (p0 * p0 + p1 * p1) + (p2 * p2 + p3 * p3);

    if (lane == 0) {               // lane 0 holds columns 0..3; need cols 0,1
      float sm0 = p0, sm1 = p1;
      if (sm0 > 0.f && sm0 <= 1.f) {
        int b = (int)ceilf(sm0 * 15.0f) - 1;
        b = b < 0 ? 0 : (b > 14 ? 14 : b);
        atomicAdd(&sC0[b], 1);
      }
      if (sm1 > 0.f && sm1 <= 1.f) {
        int b = (int)ceilf(sm1 * 15.0f) - 1;
        b = b < 0 ? 0 : (b > 14 ? 14 : b);
        atomicAdd(&sC1[b], 1);
      }
      int f0 = (int)floorf(sm0 * 15.0f); f0 = f0 < 0 ? 0 : (f0 > 14 ? 14 : f0);
      int f1 = (int)floorf(sm1 * 15.0f); f1 = f1 < 0 ? 0 : (f1 > 14 ? 14 : f1);
      atomicAdd(&sF0[f0], 1); atomicAdd(&sS0[f0], sm0);
      atomicAdd(&sF1[f1], 1); atomicAdd(&sS1[f1], sm1);
    }
  }

  #pragma unroll
  for (int off = 16; off > 0; off >>= 1) accSq += __shfl_xor(accSq, off, 32);
  if (lane == 0) sWave[waveInBlk] = accSq;
  __syncthreads();

  if (tid == 0) {
    double t = 0.0;
    #pragma unroll
    for (int w = 0; w < 8; ++w) t += (double)sWave[w];
    atomicAdd(&ws->totalSq, t);
  }
  if (tid < 15) {
    atomicAdd(&ws->cntC0[tid], sC0[tid]);
    atomicAdd(&ws->cntC1[tid], sC1[tid]);
    atomicAdd(&ws->cntF0[tid], sF0[tid]);
    atomicAdd(&ws->cntF1[tid], sF1[tid]);
    atomicAdd(&ws->sumF0[tid], sS0[tid]);
    atomicAdd(&ws->sumF1[tid], sS1[tid]);
  }
}

// Finalize (one wave): four 15-elem bin-space dot products via two
// v_wmma_f32_16x16x32_f16 using a hi/lo f16 split for f32-grade accuracy.
//   A rows:  0:16*t0  1:16*t1  2:16*t0^2  3:16*t1^2   (scaled up x16)
//   B cols:  0:S0/16  1:S1/16  2:N0/16    3:N1/16     (scaled down /16)
//   D[m][m] = d0..d3; result = (totalSq - 2(d0+d1) + d2 + d3) / (N*C)
__global__ void __launch_bounds__(32) ece_final(const Ws* __restrict__ ws,
                                                float* __restrict__ out, int nRows) {
  __shared__ float rowv[4][16];
  __shared__ float colv[4][16];
  __shared__ float sd[4];
  const int lane = threadIdx.x;

  if (lane < 16) {
    float t0 = 0.f, t1 = 0.f, S0 = 0.f, S1 = 0.f, Nf0 = 0.f, Nf1 = 0.f;
    if (lane < NB) {
      int c0 = ws->cntC0[lane];
      int c1 = ws->cntC1[lane];
      float cf0 = (float)c0, cf1 = (float)c1;
      t0 = (c0 > 0) ? (0.99f * cf0) / fmaxf(cf0, 1.0f) : 0.f;
      t1 = (c1 > 0) ? (0.01f * cf1) / fmaxf(cf1, 1.0f) : 0.f;
      S0 = ws->sumF0[lane];        S1 = ws->sumF1[lane];
      Nf0 = (float)ws->cntF0[lane]; Nf1 = (float)ws->cntF1[lane];
    }
    rowv[0][lane] = 16.0f * t0;
    rowv[1][lane] = 16.0f * t1;
    rowv[2][lane] = 16.0f * t0 * t0;
    rowv[3][lane] = 16.0f * t1 * t1;
    colv[0][lane] = S0  * 0.0625f;
    colv[1][lane] = S1  * 0.0625f;
    colv[2][lane] = Nf0 * 0.0625f;
    colv[3][lane] = Nf1 * 0.0625f;
  }
  __syncthreads();

  const int n    = lane & 15;   // A row index == B col index for this lane
  const int half = lane >> 4;
  v16h a1, a2, b1, b2;
  #pragma unroll
  for (int e = 0; e < 16; ++e) {
    // A 16x32 f16 layout: K = (e<8) ? 8*half+e : 16+8*half+(e-8)
    int ka = (e < 8) ? (8 * half + e) : (16 + 8 * half + (e - 8));
    // B 32x16 f16 layout: K = (e<8) ? 8*half+e : 16+8*half+(e-8)
    int kb = (e < 8) ? (8 * half + e) : (16 + 8 * half + (e - 8));
    // K slot -> bin index b = K & 15 (bin 15 is zero padding)
    float av = (n < 4) ? rowv[n][ka & 15] : 0.f;
    _Float16 ahi = (_Float16)av;
    _Float16 alo = (_Float16)(av - (float)ahi);
    a1[e] = ahi;                                   // hi(row) in all K slots
    a2[e] = (ka < 16) ? alo : (_Float16)0.0f;      // lo(row) in K<16 only
    float bv = (n < 4) ? colv[n][kb & 15] : 0.f;
    _Float16 bhi = (_Float16)bv;
    _Float16 blo = (_Float16)(bv - (float)bhi);
    b1[e] = (kb < 16) ? bhi : blo;                 // hi(col) | lo(col)
    b2[e] = (kb < 16) ? bhi : (_Float16)0.0f;      // hi(col) | 0
  }

  v8f acc = {};
  // acc += hi(row)*hi(col) + hi(row)*lo(col)
  acc = __builtin_amdgcn_wmma_f32_16x16x32_f16(false, a1, false, b1,
                                               (short)0, acc, false, false);
  // acc += lo(row)*hi(col)
  acc = __builtin_amdgcn_wmma_f32_16x16x32_f16(false, a2, false, b2,
                                               (short)0, acc, false, false);

  // Diagonal D[m][m], m=0..3 lives in lane m, VGPR m
  if (lane == 0) sd[0] = acc[0];
  if (lane == 1) sd[1] = acc[1];
  if (lane == 2) sd[2] = acc[2];
  if (lane == 3) sd[3] = acc[3];
  __syncthreads();

  if (lane == 0) {
    double d0 = (double)sd[0], d1 = (double)sd[1];
    double d2 = (double)sd[2], d3 = (double)sd[3];
    double total = ws->totalSq - 2.0 * (d0 + d1) + d2 + d3;
    out[0] = (float)(total / ((double)nRows * (double)N_COLS));
  }
}

extern "C" void kernel_launch(void* const* d_in, const int* in_sizes, int n_in,
                              void* d_out, int out_size, void* d_ws, size_t ws_size,
                              hipStream_t stream) {
  const float* in  = (const float*)d_in[0];   // (N, 100) f32; target unused
  float*       out = (float*)d_out;           // scalar f32
  Ws*          ws  = (Ws*)d_ws;
  const int nRows  = in_sizes[0] / N_COLS;

  ece_init<<<1, 32, 0, stream>>>(ws);
  ece_pass<<<1024, 256, 0, stream>>>(in, ws, nRows);
  ece_final<<<1, 32, 0, stream>>>(ws, out, nRows);
}
